// ChannelLinearCombo_55671366090920
// MI455X (gfx1250) — matrix operand order repaired and checked by reading, not compiled, not run on hardware
//
#include <hip/hip_runtime.h>

typedef __attribute__((ext_vector_type(16))) __bf16 v16bf;
typedef __attribute__((ext_vector_type(8)))  __bf16 v8bf;
typedef __attribute__((ext_vector_type(4)))  __bf16 v4bf;
typedef __attribute__((ext_vector_type(8)))  float  v8f;

#define NB   32      // batch
#define CIN  256
#define COUT 512
#define HW   3136    // 56*56

#define LSTR 72      // LDS row stride in bf16 elems (144 B: 16B-aligned, bank-spread)

// ---------------------------------------------------------------------------
// Pass 1: one-time split of W (512x256 fp32) into bf16 hi/lo arrays in d_ws.
// 131072 elements; 128 blocks x 256 threads x 4 elems (float4 in, v4bf out x2).
// ---------------------------------------------------------------------------
__global__ __launch_bounds__(256)
void split_w_kernel(const float* __restrict__ w,
                    __bf16* __restrict__ wHi, __bf16* __restrict__ wLo)
{
    const int i = (blockIdx.x * 256 + threadIdx.x) * 4;
    const float4 v = *(const float4*)(w + i);
    const __bf16 h0 = (__bf16)v.x, h1 = (__bf16)v.y;
    const __bf16 h2 = (__bf16)v.z, h3 = (__bf16)v.w;
    const v4bf hv = {h0, h1, h2, h3};
    const v4bf lv = {(__bf16)(v.x - (float)h0), (__bf16)(v.y - (float)h1),
                     (__bf16)(v.z - (float)h2), (__bf16)(v.w - (float)h3)};
    *(v4bf*)(wHi + i) = hv;
    *(v4bf*)(wLo + i) = lv;
}

// ---------------------------------------------------------------------------
// Pass 2: GEMM. Block: 256 threads = 8 waves; each wave does a 16(M) x 32(N)
// strip (two 16x16 tiles sharing the A fragment). Block = 128(M) x 32(N).
// X tiles are fp32->bf16(hi,lo) split ONCE per block during staging into a
// ping-pong LDS buffer (1 barrier / K-step); B fragments are straight
// ds_load_b128. A fragments: PRE=true -> bf16 global_load_b128 from the
// pre-split arrays (zero VALU); PRE=false -> in-kernel split fallback.
// Grid: (HW/32, COUT/128, NB) = (98, 4, 32).
// ---------------------------------------------------------------------------
template <bool PRE>
__global__ __launch_bounds__(256)
void chanlin_wmma_bf16x3_kernel(const float* __restrict__ x,
                                const float* __restrict__ w,
                                const __bf16* __restrict__ wHi,
                                const __bf16* __restrict__ wLo,
                                float* __restrict__ out)
{
    __shared__ __bf16 sHi[2][32 * LSTR];
    __shared__ __bf16 sLo[2][32 * LSTR];

    const int tid  = threadIdx.x;
    const int lane = tid & 31;
    const int wv   = tid >> 5;
    const int m    = lane & 15;   // A row within tile == B/D column within tile
    const int half = lane >> 4;

    const int nb = blockIdx.x * 32;                 // N base within batch image
    const int m0 = blockIdx.y * 128 + wv * 16;      // output-channel base
    const int b  = blockIdx.z;

    const float* xb = x + (size_t)b * CIN * HW;

    // staging coords: thread owns column n = tid&31, k-quad k4 = (tid>>5)*4
    const int s_n  = tid & 31;
    const int s_k4 = (tid >> 5) * 4;    // 0,4,...,28
    const float* xs0 = xb + (size_t)s_k4 * HW + nb + s_n;

    v8f acc0 = {};
    v8f acc1 = {};

    float f0, f1, f2, f3;

    // ---- prologue: load + convert + store tile kb=0 into buffer 0 ----
    f0 = xs0[0]; f1 = xs0[HW]; f2 = xs0[2 * HW]; f3 = xs0[3 * HW];
    {
        const __bf16 h0 = (__bf16)f0, h1 = (__bf16)f1;
        const __bf16 h2 = (__bf16)f2, h3 = (__bf16)f3;
        const v4bf hv = {h0, h1, h2, h3};
        const v4bf lv = {(__bf16)(f0 - (float)h0), (__bf16)(f1 - (float)h1),
                         (__bf16)(f2 - (float)h2), (__bf16)(f3 - (float)h3)};
        *(v4bf*)&sHi[0][s_n * LSTR + s_k4] = hv;
        *(v4bf*)&sLo[0][s_n * LSTR + s_k4] = lv;
    }

    int cur = 0;
    for (int kb = 0; kb < CIN; kb += 32) {
        __syncthreads();   // staged tile `cur` visible to all waves

        const bool hasNext = (kb + 32) < CIN;
        if (hasNext) {     // issue next tile's global loads before the WMMAs
            const float* xs = xs0 + (size_t)(kb + 32) * HW;
            f0 = xs[0]; f1 = xs[HW]; f2 = xs[2 * HW]; f3 = xs[3 * HW];
        }

        // ---- A fragment: K = half*8 + [0,8) and 16 + half*8 + [0,8) ----
        v16bf ahi, alo;
        if constexpr (PRE) {
            const size_t ro = (size_t)(m0 + m) * CIN + half * 8 + kb;
            const v8bf h0 = *(const v8bf*)(wHi + ro);
            const v8bf h1 = *(const v8bf*)(wHi + ro + 16);
            const v8bf l0 = *(const v8bf*)(wLo + ro);
            const v8bf l1 = *(const v8bf*)(wLo + ro + 16);
            ahi = __builtin_shufflevector(h0, h1, 0,1,2,3,4,5,6,7,
                                                  8,9,10,11,12,13,14,15);
            alo = __builtin_shufflevector(l0, l1, 0,1,2,3,4,5,6,7,
                                                  8,9,10,11,12,13,14,15);
        } else {
            const float* ap = w + (size_t)(m0 + m) * CIN + half * 8 + kb;
            const float4 a0 = *(const float4*)(ap + 0);
            const float4 a1 = *(const float4*)(ap + 4);
            const float4 a2 = *(const float4*)(ap + 16);
            const float4 a3 = *(const float4*)(ap + 20);
            float av[16];
            av[0]=a0.x;  av[1]=a0.y;  av[2]=a0.z;  av[3]=a0.w;
            av[4]=a1.x;  av[5]=a1.y;  av[6]=a1.z;  av[7]=a1.w;
            av[8]=a2.x;  av[9]=a2.y;  av[10]=a2.z; av[11]=a2.w;
            av[12]=a3.x; av[13]=a3.y; av[14]=a3.z; av[15]=a3.w;
#pragma unroll
            for (int e = 0; e < 16; ++e) {
                const __bf16 h = (__bf16)av[e];
                ahi[e] = h;
                alo[e] = (__bf16)(av[e] - (float)h);
            }
        }

        // ---- two N-tiles: B fragments via ds_load_b128 + 3 WMMAs each ----
#pragma unroll
        for (int j = 0; j < 2; ++j) {
            const int col = (j * 16 + m) * LSTR + half * 8;
            const v16bf bhi = __builtin_shufflevector(
                *(const v8bf*)&sHi[cur][col], *(const v8bf*)&sHi[cur][col + 16],
                0,1,2,3,4,5,6,7,8,9,10,11,12,13,14,15);
            const v16bf blo = __builtin_shufflevector(
                *(const v8bf*)&sLo[cur][col], *(const v8bf*)&sLo[cur][col + 16],
                0,1,2,3,4,5,6,7,8,9,10,11,12,13,14,15);
            v8f acc = j ? acc1 : acc0;
            acc = __builtin_amdgcn_wmma_f32_16x16x32_bf16(
                      false, ahi, false, bhi, (short)0, acc, false, false);
            acc = __builtin_amdgcn_wmma_f32_16x16x32_bf16(
                      false, ahi, false, blo, (short)0, acc, false, false);
            acc = __builtin_amdgcn_wmma_f32_16x16x32_bf16(
                      false, alo, false, bhi, (short)0, acc, false, false);
            if (j) acc1 = acc; else acc0 = acc;
        }

        // ---- convert + store next tile into the other buffer (no barrier:
        //      readers of that buffer finished before this iteration's sync)
        if (hasNext) {
            const int nxt = cur ^ 1;
            const __bf16 h0 = (__bf16)f0, h1 = (__bf16)f1;
            const __bf16 h2 = (__bf16)f2, h3 = (__bf16)f3;
            const v4bf hv = {h0, h1, h2, h3};
            const v4bf lv = {(__bf16)(f0 - (float)h0), (__bf16)(f1 - (float)h1),
                             (__bf16)(f2 - (float)h2), (__bf16)(f3 - (float)h3)};
            *(v4bf*)&sHi[nxt][s_n * LSTR + s_k4] = hv;
            *(v4bf*)&sLo[nxt][s_n * LSTR + s_k4] = lv;
        }
        cur ^= 1;
    }

    // ---- store D: VGPR g -> M = g + 8*half, N = lane%16 (64B coalesced rows)
#pragma unroll
    for (int j = 0; j < 2; ++j) {
        float* op = out + ((size_t)b * COUT + (size_t)(m0 + half * 8)) * HW
                        + nb + j * 16 + m;
        v8f acc = j ? acc1 : acc0;
#pragma unroll
        for (int g = 0; g < 8; ++g) {
            op[(size_t)g * HW] = acc[g];
        }
    }
}

extern "C" void kernel_launch(void* const* d_in, const int* in_sizes, int n_in,
                              void* d_out, int out_size, void* d_ws, size_t ws_size,
                              hipStream_t stream) {
    const float* x = (const float*)d_in[0];   // (32, 256, 56, 56) fp32
    const float* w = (const float*)d_in[1];   // (512, 256) fp32
    float* out = (float*)d_out;               // (32, 512, 56, 56) fp32

    dim3 grid(HW / 32, COUT / 128, NB);       // (98, 4, 32)
    dim3 block(256);

    const size_t needWs = (size_t)COUT * CIN * 2 * sizeof(__bf16);  // 512 KB
    if (ws_size >= needWs) {
        __bf16* wHi = (__bf16*)d_ws;
        __bf16* wLo = wHi + (size_t)COUT * CIN;
        hipLaunchKernelGGL(split_w_kernel, dim3((COUT * CIN) / (256 * 4)),
                           dim3(256), 0, stream, w, wHi, wLo);
        hipLaunchKernelGGL((chanlin_wmma_bf16x3_kernel<true>), grid, block, 0,
                           stream, x, w, wHi, wLo, out);
    } else {
        hipLaunchKernelGGL((chanlin_wmma_bf16x3_kernel<false>), grid, block, 0,
                           stream, x, w, (const __bf16*)nullptr,
                           (const __bf16*)nullptr, out);
    }
}
